// MultiHeadAttention_41686952575367
// MI455X (gfx1250) — matrix operand order repaired and checked by reading, not compiled, not run on hardware
//
#include <hip/hip_runtime.h>
#include <math.h>
#include <cstdint>

// MI455X / gfx1250: wave32, WMMA 16x16x32 f16 -> f32 accumulate.
// MHA: B=4, T=2048, C=1024, H=16, D=64.
// Pipeline: f32->f16 convert; 3x WMMA GEMM projections (V stored transposed);
// flash-attention with TDM (tensor_load_to_lds) double-buffered K/V blocks,
// WMMA QK^T and P*V; WMMA output projection (f32 out).
// Workspace (needs ~88MB):
//   [0,16MB) xh | [16,18) Wqh | [18,20) Wkh | [20,22) Wvh | [22,24) Woh
//   [24,40) qh [B,H,T,D] | [40,56) kh [B,H,T,D] | [56,72) vth [B,H,D,T]
//   [72,88) ctxh [B*T,C]

typedef __attribute__((ext_vector_type(16))) _Float16 v16h;
typedef __attribute__((ext_vector_type(8)))  _Float16 v8h;
typedef __attribute__((ext_vector_type(8)))  float    v8f;
typedef __attribute__((ext_vector_type(4)))  unsigned u32x4;
typedef __attribute__((ext_vector_type(8)))  int      i32x8;
typedef __attribute__((ext_vector_type(4)))  int      i32x4;

#define B_  4
#define T_  2048
#define C_  1024
#define H_  16
#define D_  64
#define NEG_INF (-__builtin_inff())

#if defined(__has_builtin)
#  if __has_builtin(__builtin_amdgcn_tensor_load_to_lds) && \
      __has_builtin(__builtin_amdgcn_s_wait_tensorcnt)
#    define HAVE_TDM 1
#  endif
#endif
#ifndef HAVE_TDM
#  define HAVE_TDM 0
#endif

// ---------------------------------------------------------------- convert
__global__ void f32_to_f16_kernel(const float* __restrict__ in,
                                  _Float16* __restrict__ out, int n) {
    int i = blockIdx.x * blockDim.x + threadIdx.x;
    if (i < n) out[i] = (_Float16)in[i];
}

// A-fragment (16x32 f16, MxK). Lanes 0-15: K={0..15}; lanes 16-31:
// K={16..31}. Elements 0-7 = low chunk, 8-15 = high chunk.
__device__ __forceinline__ v16h load_afrag(const _Float16* rowptr, int lh) {
    v8h lo = *(const v8h*)(rowptr + lh * 16);
    v8h hi = *(const v8h*)(rowptr + lh * 16 + 8);
    v16h a;
#pragma unroll
    for (int i = 0; i < 8; ++i) { a[i] = lo[i]; a[i + 8] = hi[i]; }
    return a;
}

// ------------------------------------------------- TDM 2D tile -> LDS
// D# per CDNA5 ISA. 2-byte elements, tile = tensor
// (no OOB), row stride stride0 (elements). Issued once per wave (EXEC ignored).
__device__ __forceinline__ void tdm_load_2d(unsigned lds_byte_addr, const void* gsrc,
                                            unsigned td0, unsigned td1,
                                            unsigned tile0, unsigned tile1,
                                            unsigned long long stride0) {
#if HAVE_TDM
    unsigned long long ga = (unsigned long long)(uintptr_t)gsrc;
    u32x4 g0;
    g0[0] = 1u;                                              // count=1, user D#
    g0[1] = lds_byte_addr;                                   // lds_addr[31:0]
    g0[2] = (unsigned)ga;                                    // global_addr[31:0]
    g0[3] = (unsigned)((ga >> 32) & 0x01ffffffu) | (2u << 30); // addr[56:32]|type=2
    i32x8 g1;
    g1[0] = (int)(1u << 16);                                 // data_size=1 (2B)
    g1[1] = (int)((td0 & 0xffffu) << 16);                    // tensor_dim0[15:0]
    g1[2] = (int)(((td0 >> 16) & 0xffffu) | ((td1 & 0xffffu) << 16));
    g1[3] = (int)(((td1 >> 16) & 0xffffu) | ((tile0 & 0xffffu) << 16));
    g1[4] = (int)(tile1 & 0xffffu);                          // tile_dim1, dim2=0
    g1[5] = (int)(unsigned)(stride0 & 0xffffffffu);          // dim0_stride lo
    g1[6] = (int)(unsigned)((stride0 >> 32) & 0xffffu);      // dim0_stride hi
    g1[7] = 0;
    i32x4 z4 = {0, 0, 0, 0};
#  if __clang_major__ >= 23
    i32x8 z8 = {0, 0, 0, 0, 0, 0, 0, 0};
    __builtin_amdgcn_tensor_load_to_lds(g0, g1, z4, z4, z8, 0);
#  else
    __builtin_amdgcn_tensor_load_to_lds(g0, g1, z4, z4, 0);
#  endif
#endif
}

__device__ __forceinline__ void kv_wait(bool more_pending) {
#if HAVE_TDM
    if (more_pending) __builtin_amdgcn_s_wait_tensorcnt(2);  // prior block done
    else              __builtin_amdgcn_s_wait_tensorcnt(0);
#else
    __syncthreads();
#endif
    asm volatile("" ::: "memory");   // order LDS reads after TDM completion
}

// ---------------------------------------------------------------- GEMM
// y[m,n] = sum_k X[m,k]*W[n,k] + bias[n];  M=8192, N=1024, K=1024.
// One wave per 32x64 output tile (2 A-frags share every B-frag).
// MODE 0: f16 [B,H,T,D] (Q/K). MODE 1: f16 [B,H,D,T] (V^T). MODE 2: f32 [B*T,C].
template <int MODE>
__global__ __launch_bounds__(32)
void gemm16_kernel(const _Float16* __restrict__ X, const _Float16* __restrict__ W,
                   const float* __restrict__ bias, void* __restrict__ outp) {
    const int lane = threadIdx.x;
    const int cl = lane & 15;
    const int lh = lane >> 4;
    const int m0 = blockIdx.x * 32;
    const int n0 = blockIdx.y * 64;

    v8f acc[2][4] = {};
    const _Float16* xrow0 = X + (size_t)(m0 + cl) * C_;
    const _Float16* xrow1 = X + (size_t)(m0 + 16 + cl) * C_;

    for (int k0 = 0; k0 < C_; k0 += 32) {
        v16h a0 = load_afrag(xrow0 + k0, lh);
        v16h a1 = load_afrag(xrow1 + k0, lh);
#pragma unroll
        for (int nt = 0; nt < 4; ++nt) {
            const int n = n0 + nt * 16 + cl;
            v16h bf = *(const v16h*)(W + (size_t)n * C_ + k0 + lh * 16);
            acc[0][nt] = __builtin_amdgcn_wmma_f32_16x16x32_f16(
                false, a0, false, bf, (short)0, acc[0][nt], false, false);
            acc[1][nt] = __builtin_amdgcn_wmma_f32_16x16x32_f16(
                false, a1, false, bf, (short)0, acc[1][nt], false, false);
        }
    }

#pragma unroll
    for (int nt = 0; nt < 4; ++nt) {
        const int n  = n0 + nt * 16 + cl;
        const float bn = bias[n];
        const int hh = n >> 6;
        const int d  = n & 63;
#pragma unroll
        for (int mt = 0; mt < 2; ++mt) {
#pragma unroll
            for (int r = 0; r < 8; ++r) {
                const int gm = m0 + mt * 16 + r + 8 * lh;   // global row b*T + t
                const float v = acc[mt][nt][r] + bn;
                if (MODE == 2) {
                    ((float*)outp)[(size_t)gm * C_ + n] = v;
                } else {
                    const int bb = gm >> 11;
                    const int tt = gm & (T_ - 1);
                    if (MODE == 0)
                        ((_Float16*)outp)[(((size_t)bb * H_ + hh) * T_ + tt) * D_ + d] =
                            (_Float16)v;
                    else
                        ((_Float16*)outp)[(((size_t)bb * H_ + hh) * D_ + d) * T_ + tt] =
                            (_Float16)v;
                }
            }
        }
    }
}

// ---------------------------------------------------------------- attention
// One wave per (b, h, 16-row block). Flash causal softmax. K/V s-blocks are
// staged into LDS by the Tensor Data Mover, double-buffered.
__global__ __launch_bounds__(32)
void attn_kernel(const _Float16* __restrict__ qh, const _Float16* __restrict__ kh,
                 const _Float16* __restrict__ vth, _Float16* __restrict__ ctxh) {
    const int lane = threadIdx.x;
    const int cl = lane & 15;
    const int lh = lane >> 4;
    const int t0 = blockIdx.x * 16;
    const int h  = blockIdx.y;
    const int b  = blockIdx.z;
    const size_t bh = (size_t)b * H_ + h;

    __shared__ __align__(32) _Float16 Kbuf[2][32 * 64];   // rows s, cols d
    __shared__ __align__(32) _Float16 Vbuf[2][64 * 32];   // rows d, cols s
    __shared__ __align__(32) _Float16 Pbuf[16 * 32];

    const _Float16* kbase = kh  + bh * T_ * D_;
    const _Float16* vbase = vth + bh * D_ * T_;

    // stage one 32-wide s-block (K: 32x64 contiguous rows; V^T: 64x32 strided)
    auto stage_kv = [&](int buf, int s0) {
#if HAVE_TDM
        tdm_load_2d((unsigned)(uintptr_t)&Kbuf[buf][0], kbase + (size_t)s0 * D_,
                    /*td0=*/64, /*td1=*/32, /*tile0=*/64, /*tile1=*/32,
                    /*stride0=*/64ull);
        tdm_load_2d((unsigned)(uintptr_t)&Vbuf[buf][0], vbase + s0,
                    /*td0=*/32, /*td1=*/64, /*tile0=*/32, /*tile1=*/64,
                    /*stride0=*/(unsigned long long)T_);
#else
        const v16h* src = (const v16h*)(kbase + (size_t)s0 * D_);
        v16h* dst = (v16h*)&Kbuf[buf][0];
        for (int i = lane; i < 128; i += 32) dst[i] = src[i];
        for (int r = lane; r < 64; r += 32) {
            *(v16h*)&Vbuf[buf][r * 32]      = *(const v16h*)(vbase + (size_t)r * T_ + s0);
            *(v16h*)&Vbuf[buf][r * 32 + 16] = *(const v16h*)(vbase + (size_t)r * T_ + s0 + 16);
        }
#endif
    };

    // Q A-fragments for both K-steps (d 0..31, 32..63) stay resident.
    const _Float16* qrow = qh + (bh * T_ + (t0 + cl)) * D_;
    const v16h a0 = load_afrag(qrow, lh);
    const v16h a1 = load_afrag(qrow + 32, lh);

    v8f acc[4] = {};
    float m[8], l[8];
#pragma unroll
    for (int r = 0; r < 8; ++r) { m[r] = NEG_INF; l[r] = 0.0f; }

    const float scale = 0.125f;                  // 1/sqrt(D)
    const int nS = (t0 + 15) / 32 + 1;           // causal s-blocks of 32

    stage_kv(0, 0);
    for (int j = 0; j < nS; ++j) {
        const int s0 = j * 32;
        const int buf = j & 1;
        const bool more = (j + 1 < nS);
        if (more) stage_kv(buf ^ 1, s0 + 32);
        kv_wait(more);

        const _Float16* Kb = &Kbuf[buf][0];
        const _Float16* Vb = &Vbuf[buf][0];

        // ---- scores S = Q K^T (two 16x16 tiles, K=64 => 2 wmma each)
        v8f c0 = {}, c1 = {};
        {
            v16h b00 = *(const v16h*)(Kb + (size_t)cl * 64 + lh * 16);
            c0 = __builtin_amdgcn_wmma_f32_16x16x32_f16(false, a0, false, b00,
                                                        (short)0, c0, false, false);
            v16h b01 = *(const v16h*)(Kb + (size_t)cl * 64 + 32 + lh * 16);
            c0 = __builtin_amdgcn_wmma_f32_16x16x32_f16(false, a1, false, b01,
                                                        (short)0, c0, false, false);
            v16h b10 = *(const v16h*)(Kb + (size_t)(16 + cl) * 64 + lh * 16);
            c1 = __builtin_amdgcn_wmma_f32_16x16x32_f16(false, a0, false, b10,
                                                        (short)0, c1, false, false);
            v16h b11 = *(const v16h*)(Kb + (size_t)(16 + cl) * 64 + 32 + lh * 16);
            c1 = __builtin_amdgcn_wmma_f32_16x16x32_f16(false, a1, false, b11,
                                                        (short)0, c1, false, false);
        }

        // ---- mask + scale + online softmax (row t = t0 + r + 8*lh)
        float alpha[8];
#pragma unroll
        for (int r = 0; r < 8; ++r) {
            const int t = t0 + r + 8 * lh;
            float v0 = c0[r] * scale; if (s0 + cl      > t) v0 = NEG_INF;
            float v1 = c1[r] * scale; if (s0 + 16 + cl > t) v1 = NEG_INF;
            float mx = fmaxf(v0, v1);
            mx = fmaxf(mx, __shfl_xor(mx, 1, 32));
            mx = fmaxf(mx, __shfl_xor(mx, 2, 32));
            mx = fmaxf(mx, __shfl_xor(mx, 4, 32));
            mx = fmaxf(mx, __shfl_xor(mx, 8, 32));
            const float mn = fmaxf(m[r], mx);
            alpha[r] = __expf(m[r] - mn);
            m[r] = mn;
            const float p0 = __expf(v0 - mn);
            const float p1 = __expf(v1 - mn);
            c0[r] = p0; c1[r] = p1;
            float rs = p0 + p1;
            rs += __shfl_xor(rs, 1, 32);
            rs += __shfl_xor(rs, 2, 32);
            rs += __shfl_xor(rs, 4, 32);
            rs += __shfl_xor(rs, 8, 32);
            l[r] = l[r] * alpha[r] + rs;
        }
#pragma unroll
        for (int nt = 0; nt < 4; ++nt)
#pragma unroll
            for (int r = 0; r < 8; ++r) acc[nt][r] *= alpha[r];

        // ---- transpose P (C-layout f32) -> A-layout f16 via LDS
#pragma unroll
        for (int r = 0; r < 8; ++r) {
            const int row = r + 8 * lh;
            Pbuf[row * 32 + cl]      = (_Float16)c0[r];
            Pbuf[row * 32 + 16 + cl] = (_Float16)c1[r];
        }
        __syncthreads();
        const v16h pf = load_afrag(&Pbuf[(lane & 15) * 32], lh);
        __syncthreads();

        // ---- acc += P @ V (V^T rows in LDS: contiguous 32B B-frags)
#pragma unroll
        for (int nt = 0; nt < 4; ++nt) {
            v16h vb = *(const v16h*)(Vb + (size_t)(nt * 16 + cl) * 32 + lh * 16);
            acc[nt] = __builtin_amdgcn_wmma_f32_16x16x32_f16(
                false, pf, false, vb, (short)0, acc[nt], false, false);
        }
    }

    // ---- normalize and store ctx f16 [B*T, C]
    float inv[8];
#pragma unroll
    for (int r = 0; r < 8; ++r) inv[r] = 1.0f / l[r];
#pragma unroll
    for (int nt = 0; nt < 4; ++nt) {
        const int d = nt * 16 + cl;
#pragma unroll
        for (int r = 0; r < 8; ++r) {
            const int t = t0 + r + 8 * lh;
            ctxh[((size_t)b * T_ + t) * C_ + h * D_ + d] =
                (_Float16)(acc[nt][r] * inv[r]);
        }
    }
}

// ---------------------------------------------------------------- launch
extern "C" void kernel_launch(void* const* d_in, const int* in_sizes, int n_in,
                              void* d_out, int out_size, void* d_ws, size_t ws_size,
                              hipStream_t stream) {
    const float* x  = (const float*)d_in[0];
    const float* Wq = (const float*)d_in[1];
    const float* bq = (const float*)d_in[2];
    const float* Wk = (const float*)d_in[3];
    const float* bk = (const float*)d_in[4];
    const float* Wv = (const float*)d_in[5];
    const float* bv = (const float*)d_in[6];
    const float* Wo = (const float*)d_in[7];
    const float* bo = (const float*)d_in[8];

    const size_t MB = 1024ull * 1024ull;
    char* ws = (char*)d_ws;
    _Float16* xh   = (_Float16*)(ws);
    _Float16* Wqh  = (_Float16*)(ws + 16 * MB);
    _Float16* Wkh  = (_Float16*)(ws + 18 * MB);
    _Float16* Wvh  = (_Float16*)(ws + 20 * MB);
    _Float16* Woh  = (_Float16*)(ws + 22 * MB);
    _Float16* qh   = (_Float16*)(ws + 24 * MB);
    _Float16* kh   = (_Float16*)(ws + 40 * MB);
    _Float16* vth  = (_Float16*)(ws + 56 * MB);
    _Float16* ctxh = (_Float16*)(ws + 72 * MB);

    const int nX = B_ * T_ * C_;   // 8388608
    const int nW = C_ * C_;        // 1048576
    f32_to_f16_kernel<<<(nX + 255) / 256, 256, 0, stream>>>(x, xh, nX);
    f32_to_f16_kernel<<<(nW + 255) / 256, 256, 0, stream>>>(Wq, Wqh, nW);
    f32_to_f16_kernel<<<(nW + 255) / 256, 256, 0, stream>>>(Wk, Wkh, nW);
    f32_to_f16_kernel<<<(nW + 255) / 256, 256, 0, stream>>>(Wv, Wvh, nW);
    f32_to_f16_kernel<<<(nW + 255) / 256, 256, 0, stream>>>(Wo, Woh, nW);

    dim3 gg(B_ * T_ / 32, C_ / 64);   // 256 x 16 waves, 32x64 tiles
    gemm16_kernel<0><<<gg, 32, 0, stream>>>(xh, Wqh, bq, (void*)qh);
    gemm16_kernel<0><<<gg, 32, 0, stream>>>(xh, Wkh, bk, (void*)kh);
    gemm16_kernel<1><<<gg, 32, 0, stream>>>(xh, Wvh, bv, (void*)vth);

    attn_kernel<<<dim3(T_ / 16, H_, B_), 32, 0, stream>>>(qh, kh, vth, ctxh);

    gemm16_kernel<2><<<gg, 32, 0, stream>>>(ctxh, Woh, bo, d_out);
}
